// ToolRelationshipGNN_38508676776618
// MI455X (gfx1250) — compile-verified, matrix operand order checked by reading, not run on hardware
//
#include <hip/hip_runtime.h>

// ---------------------------------------------------------------------------
// GNN forward for MI455X (gfx1250, wave32, WMMA 16x16x32 f16 -> f32 acc)
// ---------------------------------------------------------------------------
#define BB 2
#define NN 10000
#define EE 160000
#define HH 128
#define RROWS (BB * NN)   // 20000 flattened node rows

typedef __attribute__((ext_vector_type(16))) _Float16 v16h;
typedef __attribute__((ext_vector_type(8)))  _Float16 v8h;
typedef __attribute__((ext_vector_type(8)))  float    v8f;

// Load a 16x32 f16 A-fragment slice for this lane from LDS.
// p points at (row base + grp*8) in halves; lo = K[kb..kb+7], hi = K[kb+16..kb+23]
__device__ __forceinline__ v16h frag_ld(const _Float16* p) {
    v8h lo = *(const v8h*)p;
    v8h hi = *(const v8h*)(p + 16);
    return __builtin_shufflevector(lo, hi, 0,1,2,3,4,5,6,7,8,9,10,11,12,13,14,15);
}

__device__ __forceinline__ v8f wmma16(v16h a, v16h b, v8f c) {
    return __builtin_amdgcn_wmma_f32_16x16x32_f16(false, a, false, b, (short)0, c,
                                                  false, false);
}

// sum across the 16-lane half-group of a wave32
__device__ __forceinline__ float grp16_sum(float x) {
    x += __shfl_xor(x, 1, 32);
    x += __shfl_xor(x, 2, 32);
    x += __shfl_xor(x, 4, 32);
    x += __shfl_xor(x, 8, 32);
    return x;
}

__device__ __forceinline__ float sigmoidf(float x) { return 1.0f / (1.0f + expf(-x)); }

// ---------------------------------------------------------------------------
// Pack f32 weight (Nout x Kin, row-major) into f16 WMMA-B fragments.
// Fragment (nt, ks): lane l, half h  ->  W[nt*16 + (l&15)][(ks*32 + kmap(h,l>>4) + kofs) % Kin]
// kmap(h,g) = (h<8 ? h : h+8) + g*8   (matches the 16-bit A/B VGPR layout)
// ---------------------------------------------------------------------------
__global__ __launch_bounds__(256) void pack_w_kernel(const float* __restrict__ W,
                                                     _Float16* __restrict__ P,
                                                     int Nout, int Kin, int kofs) {
    int idx = blockIdx.x * 256 + threadIdx.x;
    if (idx >= Nout * Kin) return;
    int h    = idx & 15;
    int lane = (idx >> 4) & 31;
    int frag = idx >> 9;
    int ksteps = Kin >> 5;
    int nt = frag / ksteps;
    int ks = frag - nt * ksteps;
    int kk = (h < 8 ? h : h + 8) + (lane >> 4) * 8;
    int row = nt * 16 + (lane & 15);
    int col = (ks * 32 + kk + kofs) & (Kin - 1);   // Kin is 128 or 256
    P[idx] = (_Float16)W[row * Kin + col];
}

// ---------------------------------------------------------------------------
// Y = relu(LN(X @ W^T + b; g, be))   X:(R,128) f32, packed W:(8 nt x 4 ks)
// 64 rows / block, 128 threads (4 waves x 16 rows)
// ---------------------------------------------------------------------------
__global__ __launch_bounds__(128) void node_mlp_kernel(
    const float* __restrict__ X, int R,
    const _Float16* __restrict__ Wp,
    const float* __restrict__ b, const float* __restrict__ g,
    const float* __restrict__ be, float* __restrict__ Y)
{
    __shared__ __align__(16) _Float16 sA[64 * 136];
    __shared__ __align__(16) float    sO[64 * 132];
    __shared__ float sMean[64], sRstd[64];
    int tid = threadIdx.x;
    int r0 = blockIdx.x * 64;

    for (int i = tid; i < 64 * 32; i += 128) {
        int row = i >> 5, c4 = (i & 31) * 4;
        int gr = r0 + row;
        float4 v = (gr < R) ? *(const float4*)(X + (size_t)gr * 128 + c4)
                            : make_float4(0.f, 0.f, 0.f, 0.f);
        _Float16* d = sA + row * 136 + c4;
        d[0] = (_Float16)v.x; d[1] = (_Float16)v.y;
        d[2] = (_Float16)v.z; d[3] = (_Float16)v.w;
    }
    __syncthreads();

    int wave = tid >> 5, lane = tid & 31;
    int grp = lane >> 4, colb = lane & 15;
    int mrow = (wave << 4) + colb;
    int orow = (wave << 4) + grp * 8;
    const _Float16* arow = sA + mrow * 136 + grp * 8;

    // cache all A fragments in registers (loaded from LDS exactly once)
    v16h afr[4];
    #pragma unroll
    for (int ks = 0; ks < 4; ++ks) afr[ks] = frag_ld(arow + ks * 32);

    #pragma unroll
    for (int nt = 0; nt < 8; ++nt) {
        v8f acc = {};
        const v16h* wp = (const v16h*)Wp + nt * 4 * 32 + lane;
        #pragma unroll
        for (int ks = 0; ks < 4; ++ks)
            acc = wmma16(afr[ks], wp[ks * 32], acc);
        int col = nt * 16 + colb;
        float bias = b[col];
        #pragma unroll
        for (int r = 0; r < 8; ++r)
            sO[(orow + r) * 132 + col] = acc[r] + bias;
    }
    __syncthreads();

    if (tid < 64) {
        float s = 0.f, s2 = 0.f;
        const float* rp = sO + tid * 132;
        for (int c = 0; c < 128; ++c) { float v = rp[c]; s += v; s2 += v * v; }
        float m = s * (1.f / 128.f);
        sMean[tid] = m;
        sRstd[tid] = rsqrtf(s2 * (1.f / 128.f) - m * m + 1e-5f);
    }
    __syncthreads();

    for (int i = tid; i < 64 * 32; i += 128) {
        int row = i >> 5, c4 = (i & 31) * 4;
        int gr = r0 + row;
        if (gr < R) {
            float m = sMean[row], rs = sRstd[row];
            const float* rp = sO + row * 132 + c4;
            float4 o;
            o.x = fmaxf((rp[0] - m) * rs * g[c4 + 0] + be[c4 + 0], 0.f);
            o.y = fmaxf((rp[1] - m) * rs * g[c4 + 1] + be[c4 + 1], 0.f);
            o.z = fmaxf((rp[2] - m) * rs * g[c4 + 2] + be[c4 + 2], 0.f);
            o.w = fmaxf((rp[3] - m) * rs * g[c4 + 3] + be[c4 + 3], 0.f);
            *(float4*)(Y + (size_t)gr * 128 + c4) = o;
        }
    }
}

// ---------------------------------------------------------------------------
// Edge pass 1: gather hs/hd, t1 = relu(LN([hs,hd]@Wm1^T+bm1)), msg = t1@Wm2^T+bm2,
// t2 = tanh([msg,hd]@Wa1k^T+ba1) (Wa1 packed with K rotated by 128),
// ex = exp(t2.Wa2 + ba2); den[dst] += ex.   64 edges / block.
// ---------------------------------------------------------------------------
__global__ __launch_bounds__(128) void edge_pass1_kernel(
    const float* __restrict__ hcur,
    const int* __restrict__ esrc, const int* __restrict__ edst,
    const _Float16* __restrict__ Wm1p, const float* __restrict__ bm1,
    const float* __restrict__ gm1, const float* __restrict__ bem1,
    const _Float16* __restrict__ Wm2p, const float* __restrict__ bm2,
    const _Float16* __restrict__ Wa1p, const float* __restrict__ ba1,
    const float* __restrict__ Wa2, const float* __restrict__ ba2,
    _Float16* __restrict__ msg_out, float* __restrict__ ex_out,
    float* __restrict__ den_out)
{
    __shared__ __align__(16) _Float16 sA[64 * 264];   // [hs | hd] -> later [msg | hd]
    __shared__ __align__(16) _Float16 sT[64 * 136];   // t1
    __shared__ int sSrc[64], sDst[64];
    int tid = threadIdx.x;
    int b = blockIdx.y;
    int e0 = blockIdx.x * 64;
    const float* hb = hcur + (size_t)b * NN * 128;

    if (tid < 64) {
        int e = e0 + tid;
        sSrc[tid] = (e < EE) ? esrc[e] : 0;
        sDst[tid] = (e < EE) ? edst[e] : 0;
    }
    __syncthreads();

    for (int i = tid; i < 64 * 64; i += 128) {
        int row = i >> 6, q = (i & 63) * 4;
        int node = (q < 128) ? sSrc[row] : sDst[row];
        int c = q & 127;
        float4 v = *(const float4*)(hb + (size_t)node * 128 + c);
        _Float16* d = sA + row * 264 + q;
        d[0] = (_Float16)v.x; d[1] = (_Float16)v.y;
        d[2] = (_Float16)v.z; d[3] = (_Float16)v.w;
    }
    __syncthreads();

    int wave = tid >> 5, lane = tid & 31;
    int grp = lane >> 4, colb = lane & 15;
    int mrow = (wave << 4) + colb;
    int orow = (wave << 4) + grp * 8;

    float g1l[8], be1l[8], bm1l[8], bm2l[8], ba1l[8], wa2l[8];
    #pragma unroll
    for (int nt = 0; nt < 8; ++nt) {
        int c = nt * 16 + colb;
        g1l[nt] = gm1[c];  be1l[nt] = bem1[c]; bm1l[nt] = bm1[c];
        bm2l[nt] = bm2[c]; ba1l[nt] = ba1[c];  wa2l[nt] = Wa2[c];
    }

    const _Float16* arow = sA + mrow * 264 + grp * 8;

    // ---- GEMM1 (K=256) + LayerNorm + relu, all in-wave ----
    {
        v16h afr[8];
        #pragma unroll
        for (int ks = 0; ks < 8; ++ks) afr[ks] = frag_ld(arow + ks * 32);

        v8f accs[8];
        #pragma unroll
        for (int nt = 0; nt < 8; ++nt) {
            v8f acc = {};
            const v16h* wp = (const v16h*)Wm1p + nt * 8 * 32 + lane;
            #pragma unroll
            for (int ks = 0; ks < 8; ++ks)
                acc = wmma16(afr[ks], wp[ks * 32], acc);
            #pragma unroll
            for (int r = 0; r < 8; ++r) acc[r] += bm1l[nt];
            accs[nt] = acc;
        }
        #pragma unroll
        for (int r = 0; r < 8; ++r) {
            float s = 0.f, s2 = 0.f;
            #pragma unroll
            for (int nt = 0; nt < 8; ++nt) { float v = accs[nt][r]; s += v; s2 += v * v; }
            s = grp16_sum(s); s2 = grp16_sum(s2);
            float m = s * (1.f / 128.f);
            float rs = rsqrtf(s2 * (1.f / 128.f) - m * m + 1e-5f);
            #pragma unroll
            for (int nt = 0; nt < 8; ++nt) {
                float v = fmaxf((accs[nt][r] - m) * rs * g1l[nt] + be1l[nt], 0.f);
                sT[(orow + r) * 136 + nt * 16 + colb] = (_Float16)v;
            }
        }
    }
    __syncthreads();

    // ---- GEMM2 (K=128): msg -> overwrite hs half of sA ----
    {
        const _Float16* trow = sT + mrow * 136 + grp * 8;
        v16h tfr[4];
        #pragma unroll
        for (int ks = 0; ks < 4; ++ks) tfr[ks] = frag_ld(trow + ks * 32);

        #pragma unroll
        for (int nt = 0; nt < 8; ++nt) {
            v8f acc = {};
            const v16h* wp = (const v16h*)Wm2p + nt * 4 * 32 + lane;
            #pragma unroll
            for (int ks = 0; ks < 4; ++ks)
                acc = wmma16(tfr[ks], wp[ks * 32], acc);
            int col = nt * 16 + colb;
            #pragma unroll
            for (int r = 0; r < 8; ++r)
                sA[(orow + r) * 264 + col] = (_Float16)(acc[r] + bm2l[nt]);
        }
    }
    __syncthreads();

    // coalesced msg copy to global (f16)
    for (int i = tid; i < 64 * 32; i += 128) {
        int row = i >> 5, c4 = (i & 31) * 4;
        int e = e0 + row;
        if (e < EE)
            *(uint2*)(msg_out + ((size_t)b * EE + e) * 128 + c4) =
                *(const uint2*)(sA + row * 264 + c4);
    }

    // ---- GEMM3 (K=256, A=[msg|hd], Wa1 packed K-rotated) + tanh + dot Wa2 ----
    float dotp[8] = {0.f,0.f,0.f,0.f,0.f,0.f,0.f,0.f};
    {
        v16h afr[8];
        #pragma unroll
        for (int ks = 0; ks < 8; ++ks) afr[ks] = frag_ld(arow + ks * 32);

        #pragma unroll
        for (int nt = 0; nt < 8; ++nt) {
            v8f acc = {};
            const v16h* wp = (const v16h*)Wa1p + nt * 8 * 32 + lane;
            #pragma unroll
            for (int ks = 0; ks < 8; ++ks)
                acc = wmma16(afr[ks], wp[ks * 32], acc);
            #pragma unroll
            for (int r = 0; r < 8; ++r)
                dotp[r] += tanhf(acc[r] + ba1l[nt]) * wa2l[nt];
        }
    }
    float ba2v = ba2[0];
    #pragma unroll
    for (int r = 0; r < 8; ++r) {
        float ex = expf(grp16_sum(dotp[r]) + ba2v);   // |logit| <= ~2, safe
        if (colb == 0) {
            int row = orow + r;
            int e = e0 + row;
            if (e < EE) {
                ex_out[(size_t)b * EE + e] = ex;
                atomicAdd(den_out + (size_t)b * NN + sDst[row], ex);
            }
        }
    }
}

// ---------------------------------------------------------------------------
// Edge pass 2: agg[b, dst, :] += msg * (ex / den[dst])
// ---------------------------------------------------------------------------
__global__ __launch_bounds__(256) void edge_pass2_kernel(
    const _Float16* __restrict__ msg, const float* __restrict__ ex,
    const float* __restrict__ den, const int* __restrict__ edst,
    float* __restrict__ agg)
{
    size_t i = (size_t)blockIdx.x * 256 + threadIdx.x;
    if (i >= (size_t)BB * EE * 32) return;
    int c4 = (int)(i & 31) * 4;
    size_t be = i >> 5;
    int b = (int)(be / EE);
    int e = (int)(be - (size_t)b * EE);
    int d = edst[e];
    float w = ex[be] / den[(size_t)b * NN + d];
    const _Float16* mp = msg + be * 128 + c4;
    float* ap = agg + ((size_t)b * NN + d) * 128 + c4;
    atomicAdd(ap + 0, (float)mp[0] * w);
    atomicAdd(ap + 1, (float)mp[1] * w);
    atomicAdd(ap + 2, (float)mp[2] * w);
    atomicAdd(ap + 3, (float)mp[3] * w);
}

// ---------------------------------------------------------------------------
// GRU: gi = agg@Wih^T+bih, gh = h@Whh^T+bhh, gates, hnew. 64 rows/block.
// ---------------------------------------------------------------------------
__global__ __launch_bounds__(128) void gru_kernel(
    const float* __restrict__ agg, const float* __restrict__ hold,
    const _Float16* __restrict__ Wihp, const float* __restrict__ bih,
    const _Float16* __restrict__ Whhp, const float* __restrict__ bhh,
    float* __restrict__ hnew, int R)
{
    __shared__ __align__(16) _Float16 sX[64 * 136];
    __shared__ __align__(16) _Float16 sH[64 * 136];
    int tid = threadIdx.x;
    int r0 = blockIdx.x * 64;

    for (int i = tid; i < 64 * 32; i += 128) {
        int row = i >> 5, c4 = (i & 31) * 4;
        int gr = r0 + row;
        float4 x = (gr < R) ? *(const float4*)(agg + (size_t)gr * 128 + c4)
                            : make_float4(0.f, 0.f, 0.f, 0.f);
        float4 hv = (gr < R) ? *(const float4*)(hold + (size_t)gr * 128 + c4)
                             : make_float4(0.f, 0.f, 0.f, 0.f);
        _Float16* dx = sX + row * 136 + c4;
        dx[0] = (_Float16)x.x; dx[1] = (_Float16)x.y;
        dx[2] = (_Float16)x.z; dx[3] = (_Float16)x.w;
        _Float16* dh = sH + row * 136 + c4;
        dh[0] = (_Float16)hv.x; dh[1] = (_Float16)hv.y;
        dh[2] = (_Float16)hv.z; dh[3] = (_Float16)hv.w;
    }
    __syncthreads();

    int wave = tid >> 5, lane = tid & 31;
    int grp = lane >> 4, colb = lane & 15;
    int mrow = (wave << 4) + colb;
    int orow = (wave << 4) + grp * 8;
    const _Float16* xrow = sX + mrow * 136 + grp * 8;
    const _Float16* hrow = sH + mrow * 136 + grp * 8;

    // cache A fragments for both GEMM inputs once
    v16h xfr[4], hfr[4];
    #pragma unroll
    for (int ks = 0; ks < 4; ++ks) {
        xfr[ks] = frag_ld(xrow + ks * 32);
        hfr[ks] = frag_ld(hrow + ks * 32);
    }

    v8f rg[8], zg[8], ng[8];
    for (int part = 0; part < 3; ++part) {
        #pragma unroll
        for (int nt = 0; nt < 8; ++nt) {
            int NT = part * 8 + nt;
            v8f ai = {}, ah = {};
            const v16h* wi = (const v16h*)Wihp + NT * 4 * 32 + lane;
            const v16h* wh = (const v16h*)Whhp + NT * 4 * 32 + lane;
            #pragma unroll
            for (int ks = 0; ks < 4; ++ks) {
                ai = wmma16(xfr[ks], wi[ks * 32], ai);
                ah = wmma16(hfr[ks], wh[ks * 32], ah);
            }
            int col = NT * 16 + colb;
            float bi = bih[col], bh = bhh[col];
            if (part == 0) {
                #pragma unroll
                for (int r = 0; r < 8; ++r) rg[nt][r] = sigmoidf(ai[r] + bi + ah[r] + bh);
            } else if (part == 1) {
                #pragma unroll
                for (int r = 0; r < 8; ++r) zg[nt][r] = sigmoidf(ai[r] + bi + ah[r] + bh);
            } else {
                #pragma unroll
                for (int r = 0; r < 8; ++r)
                    ng[nt][r] = tanhf(ai[r] + bi + rg[nt][r] * (ah[r] + bh));
            }
        }
    }
    #pragma unroll
    for (int nt = 0; nt < 8; ++nt) {
        int col = nt * 16 + colb;
        #pragma unroll
        for (int r = 0; r < 8; ++r) {
            int gr = r0 + orow + r;
            if (gr < R) {
                float hprev = hold[(size_t)gr * 128 + col];
                float z = zg[nt][r];
                hnew[(size_t)gr * 128 + col] = (1.f - z) * ng[nt][r] + z * hprev;
            }
        }
    }
}

// ---------------------------------------------------------------------------
// out = h @ Wo^T + bo
// ---------------------------------------------------------------------------
__global__ __launch_bounds__(128) void out_kernel(
    const float* __restrict__ hfin, const _Float16* __restrict__ Wop,
    const float* __restrict__ bo, float* __restrict__ out, int R)
{
    __shared__ __align__(16) _Float16 sA[64 * 136];
    int tid = threadIdx.x;
    int r0 = blockIdx.x * 64;
    for (int i = tid; i < 64 * 32; i += 128) {
        int row = i >> 5, c4 = (i & 31) * 4;
        int gr = r0 + row;
        float4 v = (gr < R) ? *(const float4*)(hfin + (size_t)gr * 128 + c4)
                            : make_float4(0.f, 0.f, 0.f, 0.f);
        _Float16* d = sA + row * 136 + c4;
        d[0] = (_Float16)v.x; d[1] = (_Float16)v.y;
        d[2] = (_Float16)v.z; d[3] = (_Float16)v.w;
    }
    __syncthreads();
    int wave = tid >> 5, lane = tid & 31;
    int grp = lane >> 4, colb = lane & 15;
    const _Float16* arow = sA + ((wave << 4) + colb) * 136 + grp * 8;
    int orow = (wave << 4) + grp * 8;

    v16h afr[4];
    #pragma unroll
    for (int ks = 0; ks < 4; ++ks) afr[ks] = frag_ld(arow + ks * 32);

    #pragma unroll
    for (int nt = 0; nt < 8; ++nt) {
        v8f acc = {};
        const v16h* wp = (const v16h*)Wop + nt * 4 * 32 + lane;
        #pragma unroll
        for (int ks = 0; ks < 4; ++ks)
            acc = wmma16(afr[ks], wp[ks * 32], acc);
        int col = nt * 16 + colb;
        float bias = bo[col];
        #pragma unroll
        for (int r = 0; r < 8; ++r) {
            int gr = r0 + orow + r;
            if (gr < R) out[(size_t)gr * 128 + col] = acc[r] + bias;
        }
    }
}

// ---------------------------------------------------------------------------
extern "C" void kernel_launch(void* const* d_in, const int* in_sizes, int n_in,
                              void* d_out, int out_size, void* d_ws, size_t ws_size,
                              hipStream_t stream) {
    (void)in_sizes; (void)n_in; (void)out_size; (void)ws_size;
    const float* emb = (const float*)d_in[0];
    const int* esrc  = (const int*)d_in[1];
    const int* edst  = (const int*)d_in[2];
    const float* Wn1 = (const float*)d_in[3];
    const float* bn1 = (const float*)d_in[4];
    const float* gn1 = (const float*)d_in[5];
    const float* ben1= (const float*)d_in[6];
    const float* Wn2 = (const float*)d_in[7];
    const float* bn2 = (const float*)d_in[8];
    const float* gn2 = (const float*)d_in[9];
    const float* ben2= (const float*)d_in[10];
    const float* Wm1 = (const float*)d_in[11];
    const float* bm1 = (const float*)d_in[12];
    const float* gm1 = (const float*)d_in[13];
    const float* bem1= (const float*)d_in[14];
    const float* Wm2 = (const float*)d_in[15];
    const float* bm2 = (const float*)d_in[16];
    const float* Wa1 = (const float*)d_in[17];
    const float* ba1 = (const float*)d_in[18];
    const float* Wa2 = (const float*)d_in[19];
    const float* ba2 = (const float*)d_in[20];
    const float* Wih = (const float*)d_in[21];
    const float* bih = (const float*)d_in[22];
    const float* Whh = (const float*)d_in[23];
    const float* bhh = (const float*)d_in[24];
    const float* Wo  = (const float*)d_in[25];
    const float* bo  = (const float*)d_in[26];

    char* ws = (char*)d_ws;
    size_t off = 0;
    auto take = [&](size_t bytes) -> char* {
        char* p = ws + off;
        off += (bytes + 255) & ~(size_t)255;
        return p;
    };
    _Float16* Wn1p = (_Float16*)take(128 * 128 * 2);
    _Float16* Wn2p = (_Float16*)take(128 * 128 * 2);
    _Float16* Wm1p = (_Float16*)take(128 * 256 * 2);
    _Float16* Wm2p = (_Float16*)take(128 * 128 * 2);
    _Float16* Wa1p = (_Float16*)take(128 * 256 * 2);
    _Float16* Wihp = (_Float16*)take(384 * 128 * 2);
    _Float16* Whhp = (_Float16*)take(384 * 128 * 2);
    _Float16* Wop  = (_Float16*)take(128 * 128 * 2);
    float* hA  = (float*)take((size_t)RROWS * 128 * 4);
    float* hB  = (float*)take((size_t)RROWS * 128 * 4);
    float* agg = (float*)take((size_t)RROWS * 128 * 4);
    float* den = (float*)take((size_t)RROWS * 4);
    float* ex  = (float*)take((size_t)BB * EE * 4);
    _Float16* msg = (_Float16*)take((size_t)BB * EE * 128 * 2);

    // pack weights (Wa1 with K rotated by 128 so A=[msg|hd] works)
    pack_w_kernel<<<(128*128 + 255)/256, 256, 0, stream>>>(Wn1, Wn1p, 128, 128, 0);
    pack_w_kernel<<<(128*128 + 255)/256, 256, 0, stream>>>(Wn2, Wn2p, 128, 128, 0);
    pack_w_kernel<<<(128*256 + 255)/256, 256, 0, stream>>>(Wm1, Wm1p, 128, 256, 0);
    pack_w_kernel<<<(128*128 + 255)/256, 256, 0, stream>>>(Wm2, Wm2p, 128, 128, 0);
    pack_w_kernel<<<(128*256 + 255)/256, 256, 0, stream>>>(Wa1, Wa1p, 128, 256, 128);
    pack_w_kernel<<<(384*128 + 255)/256, 256, 0, stream>>>(Wih, Wihp, 384, 128, 0);
    pack_w_kernel<<<(384*128 + 255)/256, 256, 0, stream>>>(Whh, Whhp, 384, 128, 0);
    pack_w_kernel<<<(128*128 + 255)/256, 256, 0, stream>>>(Wo,  Wop,  128, 128, 0);

    const int NODE_BLOCKS = (RROWS + 63) / 64;   // 313

    // node MLP (2 layers)
    node_mlp_kernel<<<NODE_BLOCKS, 128, 0, stream>>>(emb, RROWS, Wn1p, bn1, gn1, ben1, hA);
    node_mlp_kernel<<<NODE_BLOCKS, 128, 0, stream>>>(hA, RROWS, Wn2p, bn2, gn2, ben2, hB);

    float* cur = hB;
    float* nxt = hA;
    for (int round = 0; round < 3; ++round) {
        hipMemsetAsync(den, 0, (size_t)RROWS * 4, stream);
        hipMemsetAsync(agg, 0, (size_t)RROWS * 128 * 4, stream);

        edge_pass1_kernel<<<dim3(EE / 64, BB), 128, 0, stream>>>(
            cur, esrc, edst, Wm1p, bm1, gm1, bem1, Wm2p, bm2,
            Wa1p, ba1, Wa2, ba2, msg, ex, den);

        size_t p2 = (size_t)BB * EE * 32;
        edge_pass2_kernel<<<(unsigned)((p2 + 255) / 256), 256, 0, stream>>>(
            msg, ex, den, edst, agg);

        gru_kernel<<<NODE_BLOCKS, 128, 0, stream>>>(
            agg, cur, Wihp, bih, Whhp, bhh, nxt, RROWS);

        float* t = cur; cur = nxt; nxt = t;
    }

    out_kernel<<<NODE_BLOCKS, 128, 0, stream>>>(cur, Wop, bo, (float*)d_out, RROWS);
}
